// CasualAttention_56057913147592
// MI455X (gfx1250) — matrix operand order repaired and checked by reading, not compiled, not run on hardware
//
#include <hip/hip_runtime.h>

// Problem constants (from reference): B,T,D,H,E
#define B_ 4
#define T_ 2048
#define D_ 2048
#define H_ 16
#define E_ 128

typedef __bf16 bf16_t;
typedef __attribute__((ext_vector_type(16))) __bf16 v16bf;
typedef __attribute__((ext_vector_type(8)))  __bf16 v8bf;
typedef __attribute__((ext_vector_type(4)))  __bf16 v4bf;
typedef __attribute__((ext_vector_type(8)))  float  v8f;

union Frag { v16bf v; v8bf h[2]; };

__device__ __forceinline__ v8f wmma_bf16(const Frag& a, const Frag& b, v8f c) {
  // 8 args: (neg_a, A, neg_b, B, c_mod, C, reuse_a, reuse_b)
  return __builtin_amdgcn_wmma_f32_16x16x32_bf16(false, a.v, false, b.v,
                                                 (short)0, c, false, false);
}

// Async global -> LDS copy of 16 bytes per lane (CDNA5 TENSOR/ASYNC path,
// tracked with ASYNCcnt; no VGPR round trip).  GV addressing mode.
__device__ __forceinline__ void async_copy16(unsigned lds_off, const void* gaddr) {
  asm volatile("global_load_async_to_lds_b128 %0, %1, off"
               :: "v"(lds_off), "v"((unsigned long long)(size_t)gaddr)
               : "memory");
}
#define WAIT_ASYNC(N) asm volatile("s_wait_asynccnt " #N ::: "memory")

// ---------------------------------------------------------------------------
// Kernel 1: QKV projection.  out = x @ W^T  (both operands K-contiguous).
// grid = (D/64, (B*T)/128, 3);  z selects Wq/Wk/Wv and destination layout.
// Q,K stored [b][h][t][e] bf16; V stored transposed [b][h][e][t] bf16.
// Software-pipelined: next tile's global loads overlap this tile's WMMAs.
// ---------------------------------------------------------------------------
__global__ __launch_bounds__(256)
void qkv_proj_kernel(const float* __restrict__ x,
                     const float* __restrict__ Wq,
                     const float* __restrict__ Wk,
                     const float* __restrict__ Wv,
                     bf16_t* __restrict__ qws,
                     bf16_t* __restrict__ kws,
                     bf16_t* __restrict__ vtws)
{
  __shared__ __attribute__((aligned(16))) bf16_t As[128][48];  // 128 rows x 32 K (+pad)
  __shared__ __attribute__((aligned(16))) bf16_t Bs[64][48];   // 64 cols  x 32 K (+pad)

  const int z = blockIdx.z;
  const float* W = (z == 0) ? Wq : (z == 1) ? Wk : Wv;

  const int tid  = threadIdx.x;
  const int lane = tid & 31;
  const int w    = tid >> 5;     // wave 0..7
  const int wm   = w >> 1;       // 0..3 (row group)
  const int wn   = w & 1;        // 0..1 (col group)
  const int lh   = lane >> 4;    // lane half (wave32)
  const int lm   = lane & 15;

  const int m0 = blockIdx.y * 128;   // global row base (b*T + t)
  const int n0 = blockIdx.x * 64;    // global col base (d = h*E + e)

  const float4* xg = (const float4*)x;
  const float4* Wg = (const float4*)W;

  v8f acc[2][2] = {};
  float4 ar[4], br[2];

  // Prologue: issue tile-0 loads (all in flight before any use).
#pragma unroll
  for (int i = 0; i < 4; ++i) {
    int idx = i * 256 + tid;
    ar[i] = xg[((size_t)(m0 + (idx >> 3)) * D_) / 4 + (idx & 7)];
  }
#pragma unroll
  for (int i = 0; i < 2; ++i) {
    int idx = i * 256 + tid;
    br[i] = Wg[((size_t)(n0 + (idx >> 3)) * D_) / 4 + (idx & 7)];
  }

  for (int k0 = 0; k0 < D_; k0 += 32) {
    // Convert staged registers -> bf16 LDS tiles.
#pragma unroll
    for (int i = 0; i < 4; ++i) {
      int idx = i * 256 + tid;
      v4bf hv = {(bf16_t)ar[i].x, (bf16_t)ar[i].y, (bf16_t)ar[i].z, (bf16_t)ar[i].w};
      *(v4bf*)&As[idx >> 3][(idx & 7) * 4] = hv;
    }
#pragma unroll
    for (int i = 0; i < 2; ++i) {
      int idx = i * 256 + tid;
      v4bf hv = {(bf16_t)br[i].x, (bf16_t)br[i].y, (bf16_t)br[i].z, (bf16_t)br[i].w};
      *(v4bf*)&Bs[idx >> 3][(idx & 7) * 4] = hv;
    }
    __syncthreads();

    // Issue next tile's global loads; they overlap the WMMAs below.
    if (k0 + 32 < D_) {
#pragma unroll
      for (int i = 0; i < 4; ++i) {
        int idx = i * 256 + tid;
        ar[i] = xg[((size_t)(m0 + (idx >> 3)) * D_ + k0 + 32) / 4 + (idx & 7)];
      }
#pragma unroll
      for (int i = 0; i < 2; ++i) {
        int idx = i * 256 + tid;
        br[i] = Wg[((size_t)(n0 + (idx >> 3)) * D_ + k0 + 32) / 4 + (idx & 7)];
      }
    }

    // A frag: lane holds row m=lm; K = {8*lh..8*lh+7, 16+8*lh..16+8*lh+7}
    // B frag: lane holds col n=lm; K = {8*lh..8*lh+7, 16+8*lh..16+8*lh+7}
    const int ab = 8 * lh;
    const int bb = 16 * lh;
    Frag af[2], bfr[2];
#pragma unroll
    for (int mt = 0; mt < 2; ++mt) {
      int m = wm * 32 + mt * 16 + lm;
      af[mt].h[0] = *(const v8bf*)&As[m][ab];
      af[mt].h[1] = *(const v8bf*)&As[m][16 + ab];
    }
#pragma unroll
    for (int nt = 0; nt < 2; ++nt) {
      int n = wn * 32 + nt * 16 + lm;
      bfr[nt].h[0] = *(const v8bf*)&Bs[n][ab];
      bfr[nt].h[1] = *(const v8bf*)&Bs[n][16 + ab];
    }
#pragma unroll
    for (int mt = 0; mt < 2; ++mt)
#pragma unroll
      for (int nt = 0; nt < 2; ++nt)
        acc[mt][nt] = wmma_bf16(af[mt], bfr[nt], acc[mt][nt]);
    __syncthreads();
  }

  // Epilogue: C/D layout lane holds col n=lm, rows m = v + 8*lh.
#pragma unroll
  for (int mt = 0; mt < 2; ++mt)
#pragma unroll
    for (int nt = 0; nt < 2; ++nt)
#pragma unroll
      for (int v = 0; v < 8; ++v) {
        int gr = m0 + wm * 32 + mt * 16 + v + 8 * lh;   // b*T + t
        int gd = n0 + wn * 32 + nt * 16 + lm;           // h*E + e
        int bb_ = gr >> 11, tt = gr & (T_ - 1);
        int hh  = gd >> 7,  ee = gd & (E_ - 1);
        bf16_t val = (bf16_t)acc[mt][nt][v];
        if (z == 0)
          qws[(((size_t)bb_ * H_ + hh) * T_ + tt) * E_ + ee] = val;
        else if (z == 1)
          kws[(((size_t)bb_ * H_ + hh) * T_ + tt) * E_ + ee] = val;
        else
          vtws[(((size_t)bb_ * H_ + hh) * E_ + ee) * T_ + tt] = val;
      }
}

// ---------------------------------------------------------------------------
// Kernel 2: flash attention.  grid = (T/128, H, B), 256 threads (8 waves).
// Each wave owns 16 q-rows; online softmax in fp32; S and P*V via bf16 WMMA.
// K/V blocks double-buffered via async global->LDS copies (ASYNCcnt).
// ---------------------------------------------------------------------------
__global__ __launch_bounds__(256)
void flash_attn_kernel(const bf16_t* __restrict__ qws,
                       const bf16_t* __restrict__ kws,
                       const bf16_t* __restrict__ vtws,
                       float* __restrict__ out)
{
  __shared__ __attribute__((aligned(16))) bf16_t Qs[128][144];      // 128 q-rows x 128 e
  __shared__ __attribute__((aligned(16))) bf16_t Ks[2][32][144];    // 32 keys x 128 e, 2 bufs
  __shared__ __attribute__((aligned(16))) bf16_t Vs[2][128][48];    // 128 e x 32 keys, 2 bufs
  __shared__ __attribute__((aligned(16))) bf16_t Ps[8][16][48];     // per-wave P bounce

  const int tid  = threadIdx.x;
  const int lane = tid & 31;
  const int w    = tid >> 5;
  const int lh   = lane >> 4;
  const int lm   = lane & 15;

  const int q0 = blockIdx.x * 128;
  const int hh = blockIdx.y;
  const int bb = blockIdx.z;
  const size_t base = ((size_t)bb * H_ + hh) * (size_t)T_ * E_;  // q/k/vt all same size

  // Per-thread async copy plan: 2x16B for K block, 2x16B for Vt block.
  const int krow = tid >> 4, kc8 = tid & 15;            // idx = i*256+tid pattern below
  auto issue_kv = [&](int kb, int buf) {
#pragma unroll
    for (int i = 0; i < 2; ++i) {
      int idx = i * 256 + tid;
      int row = idx >> 4, c8 = idx & 15;                // 32 keys x 16 chunks
      async_copy16((unsigned)(size_t)&Ks[buf][row][c8 * 8],
                   &kws[base + (size_t)(kb * 32 + row) * E_ + c8 * 8]);
    }
#pragma unroll
    for (int i = 0; i < 2; ++i) {
      int idx = i * 256 + tid;
      int row = idx >> 2, c8 = idx & 3;                 // 128 e x 4 chunks
      async_copy16((unsigned)(size_t)&Vs[buf][row][c8 * 8],
                   &vtws[base + (size_t)row * T_ + kb * 32 + c8 * 8]);
    }
  };
  (void)krow; (void)kc8;

  // Stage Q tile (8 x 16B per thread) + K/V block 0, all async.
#pragma unroll
  for (int i = 0; i < 8; ++i) {
    int idx = i * 256 + tid;
    int row = idx >> 4, c8 = idx & 15;
    async_copy16((unsigned)(size_t)&Qs[row][c8 * 8],
                 &qws[base + (size_t)(q0 + row) * E_ + c8 * 8]);
  }
  issue_kv(0, 0);
  WAIT_ASYNC(4);          // in-order: my 8 Q copies done (block-0 K/V may be in flight)
  __syncthreads();

  // Pre-stage this wave's Q A-fragments for the 4 e-chunks of 32.
  const int ab = 8 * lh;
  Frag qf[4];
#pragma unroll
  for (int ek = 0; ek < 4; ++ek) {
    qf[ek].h[0] = *(const v8bf*)&Qs[w * 16 + lm][ek * 32 + ab];
    qf[ek].h[1] = *(const v8bf*)&Qs[w * 16 + lm][ek * 32 + 16 + ab];
  }

  v8f o[8] = {};
  float mrow[8], lrow[8];
#pragma unroll
  for (int v = 0; v < 8; ++v) { mrow[v] = -1e30f; lrow[v] = 0.0f; }

  const float sscale = 0.08838834764831845f;  // 1/sqrt(E)

  for (int kb = 0; kb < T_ / 32; ++kb) {
    const int cur = kb & 1;
    // Prefetch next block into the other buffer, then wait for current block
    // (in-order ASYNCcnt: <=4 outstanding means current block's copies done).
    if (kb + 1 < T_ / 32) {
      issue_kv(kb + 1, cur ^ 1);
      WAIT_ASYNC(4);
    } else {
      WAIT_ASYNC(0);
    }
    __syncthreads();   // all waves' current-block copies visible

    // S = Q * K^T for 16 q-rows x 32 keys (two 16x16 tiles, 4 e-chunks each).
    v8f s[2] = {};
#pragma unroll
    for (int kt = 0; kt < 2; ++kt) {
      int n = kt * 16 + lm;
#pragma unroll
      for (int ek = 0; ek < 4; ++ek) {
        Frag kf;
        kf.h[0] = *(const v8bf*)&Ks[cur][n][ek * 32 + ab];
        kf.h[1] = *(const v8bf*)&Ks[cur][n][ek * 32 + 16 + ab];
        s[kt] = wmma_bf16(qf[ek], kf, s[kt]);
      }
    }

    // Online softmax. Row (v + 8*lh) lives in VGPR v across one 16-lane half,
    // so xor-butterfly over {1,2,4,8} reduces within the half (wave32).
    float p0[8], p1[8], corr[8];
#pragma unroll
    for (int v = 0; v < 8; ++v) {
      float s0 = s[0][v] * sscale, s1 = s[1][v] * sscale;
      float r = fmaxf(s0, s1);
      r = fmaxf(r, __shfl_xor(r, 1));
      r = fmaxf(r, __shfl_xor(r, 2));
      r = fmaxf(r, __shfl_xor(r, 4));
      r = fmaxf(r, __shfl_xor(r, 8));
      float mn = fmaxf(mrow[v], r);
      float c  = __expf(mrow[v] - mn);
      mrow[v] = mn;
      corr[v] = c;
      float e0 = __expf(s0 - mn), e1 = __expf(s1 - mn);
      float rs = e0 + e1;
      rs += __shfl_xor(rs, 1);
      rs += __shfl_xor(rs, 2);
      rs += __shfl_xor(rs, 4);
      rs += __shfl_xor(rs, 8);
      lrow[v] = lrow[v] * c + rs;
      p0[v] = e0; p1[v] = e1;
    }
#pragma unroll
    for (int j = 0; j < 8; ++j)
#pragma unroll
      for (int v = 0; v < 8; ++v)
        o[j][v] *= corr[v];

    // Reshape P (C-layout) -> A-layout via per-wave LDS bounce.
#pragma unroll
    for (int v = 0; v < 8; ++v) {
      int m = v + 8 * lh;
      Ps[w][m][lm]      = (bf16_t)p0[v];
      Ps[w][m][16 + lm] = (bf16_t)p1[v];
    }
    Frag pf;  // same-wave LDS ordering guarantees visibility
    pf.h[0] = *(const v8bf*)&Ps[w][lm][ab];
    pf.h[1] = *(const v8bf*)&Ps[w][lm][16 + ab];

    // O += P @ V  (8 e-subtiles of 16).
#pragma unroll
    for (int j = 0; j < 8; ++j) {
      Frag vf;
      int n = j * 16 + lm;
      vf.h[0] = *(const v8bf*)&Vs[cur][n][ab];
      vf.h[1] = *(const v8bf*)&Vs[cur][n][16 + ab];
      o[j] = wmma_bf16(pf, vf, o[j]);
    }
    __syncthreads();   // everyone done reading buf `cur` before it is refilled
  }

  // Normalize and write O (fp32, [b][h][t][e]).
#pragma unroll
  for (int v = 0; v < 8; ++v) {
    float inv = 1.0f / lrow[v];
    int m = v + 8 * lh;
    size_t rowBase = (((size_t)bb * H_ + hh) * T_ + q0 + w * 16 + m) * (size_t)E_;
#pragma unroll
    for (int j = 0; j < 8; ++j)
      out[rowBase + j * 16 + lm] = o[j][v] * inv;
  }
}

// ---------------------------------------------------------------------------
extern "C" void kernel_launch(void* const* d_in, const int* in_sizes, int n_in,
                              void* d_out, int out_size, void* d_ws, size_t ws_size,
                              hipStream_t stream) {
  (void)in_sizes; (void)n_in; (void)out_size; (void)ws_size;
  const float* x  = (const float*)d_in[0];
  const float* Wq = (const float*)d_in[1];
  const float* Wk = (const float*)d_in[2];
  const float* Wv = (const float*)d_in[3];

  const size_t per = (size_t)B_ * H_ * T_ * E_;  // elements per q/k/v tensor
  bf16_t* qws  = (bf16_t*)d_ws;
  bf16_t* kws  = qws + per;
  bf16_t* vtws = kws + per;

  dim3 pg(D_ / 64, (B_ * T_) / 128, 3);
  qkv_proj_kernel<<<pg, dim3(256), 0, stream>>>(x, Wq, Wk, Wv, qws, kws, vtws);

  dim3 ag(T_ / 128, H_, B_);
  flash_attn_kernel<<<ag, dim3(256), 0, stream>>>(qws, kws, vtws, (float*)d_out);
}
